// SpectralConv1d_51857435131963
// MI455X (gfx1250) — compile-verified, matrix operand order checked by reading, not run on hardware
//
#include <hip/hip_runtime.h>
#include <hip/hip_bf16.h>

// SpectralConv1d on gfx1250: FFT truncated to 64 modes == three dense GEMMs.
// Phase 1: forward truncated DFT   (WMMA f16->f32, async-LDS double buffer)
// Phase 2: complex mode mixing     (VALU, tiny)
// Phase 3: inverse truncated rDFT  (WMMA f16->f32, async-LDS basis staging)

#define B_   16
#define L_   4096
#define H_   8
#define E_   64
#define EO_  64
#define M_   64
#define K2_  128              // concat(re,im) K dimension

typedef __attribute__((ext_vector_type(16))) _Float16 v16h;
typedef __attribute__((ext_vector_type(8)))  _Float16 v8h;
typedef __attribute__((ext_vector_type(8)))  float    v8f;

static __device__ __forceinline__ v16h mk16(v8h lo, v8h hi) {
    v16h r;
#pragma unroll
    for (int i = 0; i < 8; ++i) { r[i] = lo[i]; r[8 + i] = hi[i]; }
    return r;
}

// ---- gfx1250 async global->LDS (ASYNCcnt-tracked), GV addressing mode ----
static __device__ __forceinline__ void async_ld_b32(unsigned lds_addr, const void* gaddr) {
    asm volatile("global_load_async_to_lds_b32 %0, %1, off"
                 :: "v"(lds_addr), "v"(gaddr) : "memory");
}
static __device__ __forceinline__ void async_ld_b128(unsigned lds_addr, const void* gaddr) {
    asm volatile("global_load_async_to_lds_b128 %0, %1, off"
                 :: "v"(lds_addr), "v"(gaddr) : "memory");
}
static __device__ __forceinline__ void wait_async0() {
    asm volatile("s_wait_asynccnt 0x0" ::: "memory");
}
static __device__ __forceinline__ unsigned lds_off(const void* p) {
    return (unsigned)(unsigned long long)p;   // low 32 bits = LDS offset
}

// ---------------------------------------------------------------------------
// Basis tables (recomputed each launch; deterministic, ~0.5M sincos).
// fb[j][l] (j<64: cos, j>=64: -sin)        -> forward DFT, B-operand layout
// ib[l][k] (k<64: c_m*cos, k>=64:-c_m*sin) -> inverse DFT, B-operand layout
// ---------------------------------------------------------------------------
__global__ void spectral_basis_kernel(_Float16* __restrict__ fb,
                                      _Float16* __restrict__ ib) {
    const float TWOPI = 6.28318530717958647692f;
    int idx = blockIdx.x * blockDim.x + threadIdx.x;   // 128*4096 threads
    int l = idx & (L_ - 1);
    int j = idx >> 12;                                 // 0..127
    int m = j & 63;
    int ph = (m * l) & (L_ - 1);                       // exact angle mod 2pi
    float theta = TWOPI * (float)ph / (float)L_;
    float s, c;
    __sincosf(theta, &s, &c);
    float fwd = (j < 64) ? c : -s;
    fb[(size_t)j * L_ + l] = (_Float16)fwd;
    float cm = ((m == 0) ? 1.0f : 2.0f) / (float)L_;
    float inv = (j < 64) ? (cm * c) : (-cm * s);
    ib[(size_t)l * K2_ + j] = (_Float16)inv;
}

// ---------------------------------------------------------------------------
// Phase 1: xft[b*512+he][k] = sum_l x[b,l,he] * fb[k][l]   (k = 0..127)
// Grid: 16 batches x 8 he-blocks of 64. Block: 256 thr = 8 waves.
// Wave (r=w%4, cg=w/4): rows he0+16r..+15, cols 64*cg..+63 (4 C tiles).
// x chunk [32 l][64 he] transposed DURING the async copy into padded LDS
// ([he][l], stride 36 dw -> 16B-aligned runs, conflict-free banks).
// Double-buffered: async copy of tile i+1 overlaps WMMA on tile i.
// ---------------------------------------------------------------------------
#define XS_ 36
__global__ void spectral_dft_kernel(const float* __restrict__ x,
                                    const _Float16* __restrict__ fb,
                                    float* __restrict__ xft) {
    __shared__ float tile[2][64 * XS_];
    const int wg    = blockIdx.x;
    const int b     = wg >> 3;
    const int he0   = (wg & 7) * 64;
    const int tid   = threadIdx.x;
    const int wave  = tid >> 5;
    const int lane  = tid & 31;
    const int lid   = lane & 15;
    const int kb    = (lane >> 4) * 8;
    const int r     = wave & 3;
    const int cg    = wave >> 2;
    const int rowhe = r * 16 + lid;

    const float* xb  = x + (size_t)b * L_ * 512;
    const int    sli = tid >> 6;     // staging l sub-index (0..3)
    const int    shе = tid & 63;     // staging he index

    // prologue: async-stage tile 0 (transposing scatter, one B32 per lane op)
#pragma unroll
    for (int p = 0; p < 8; ++p) {
        int li = p * 4 + sli;
        async_ld_b32(lds_off(&tile[0][shе * XS_ + li]),
                     xb + (size_t)li * 512 + he0 + shе);
    }
    wait_async0();
    __syncthreads();

    v8f acc[4] = {};

    for (int l0 = 0; l0 < L_; l0 += 32) {
        const int cur = (l0 >> 5) & 1;
        if (l0 + 32 < L_) {          // async-prefetch next tile into other buf
#pragma unroll
            for (int p = 0; p < 8; ++p) {
                int li = p * 4 + sli;
                async_ld_b32(lds_off(&tile[cur ^ 1][shе * XS_ + li]),
                             xb + (size_t)(l0 + 32 + li) * 512 + he0 + shе);
            }
            __builtin_prefetch(fb + (size_t)(cg * 64 + lid) * L_ + l0 + 32, 0, 1);
        }

        // A fragment: row = he (lane%16), K = local l per ISA striping
        v16h a;
        const float* arow = &tile[cur][rowhe * XS_ + kb];
#pragma unroll
        for (int i = 0; i < 8; ++i) a[i]     = (_Float16)arow[i];
#pragma unroll
        for (int i = 0; i < 8; ++i) a[8 + i] = (_Float16)arow[16 + i];

#pragma unroll
        for (int t = 0; t < 4; ++t) {
            int j = cg * 64 + t * 16 + lid;            // basis row = out col
            const _Float16* brow = fb + (size_t)j * L_ + l0 + kb;
            v16h bf = mk16(*(const v8h*)brow, *(const v8h*)(brow + 16));
            acc[t] = __builtin_amdgcn_wmma_f32_16x16x32_f16(
                false, a, false, bf, (short)0, acc[t], false, false);
        }

        wait_async0();
        __syncthreads();
    }

#pragma unroll
    for (int t = 0; t < 4; ++t) {
        int ncol = cg * 64 + t * 16 + lid;
#pragma unroll
        for (int vg = 0; vg < 8; ++vg) {
            int row = r * 16 + vg + (lane >> 4) * 8;
            xft[((size_t)(b * 512 + he0 + row)) * K2_ + ncol] = acc[t][vg];
        }
    }
}

// ---------------------------------------------------------------------------
// Phase 2: complex mode mix, out_m[b,h,o,m] = sum_e Xft[b,h,e,m]*W[h,e,o,m].
// 0.14 GFLOP / 8 MB: plain VALU; writes f16 in WMMA-A-friendly [row][128].
// ---------------------------------------------------------------------------
__global__ void spectral_mix_kernel(const float* __restrict__ xft,
                                    const float* __restrict__ wr,
                                    const float* __restrict__ wi,
                                    _Float16* __restrict__ outm) {
    int idx = blockIdx.x * blockDim.x + threadIdx.x;   // B*H*EO*M threads
    int m = idx & 63;
    int o = (idx >> 6) & 63;
    int h = (idx >> 12) & 7;
    int b = idx >> 15;
    const float* xr  = xft + ((size_t)(b * 512 + h * 64)) * K2_ + m;
    const float* wrp = wr + (((size_t)h * 64) * 64 + o) * 64 + m;
    const float* wip = wi + (((size_t)h * 64) * 64 + o) * 64 + m;
    float are = 0.f, aim = 0.f;
#pragma unroll 4
    for (int e = 0; e < 64; ++e) {
        float xre = xr[e * K2_];
        float xim = xr[e * K2_ + 64];
        float wre = wrp[(size_t)e * 4096];
        float wim = wip[(size_t)e * 4096];
        are += xre * wre - xim * wim;
        aim += xre * wim + xim * wre;
    }
    size_t row = (size_t)(b * 512 + h * 64 + o);
    outm[row * K2_ + m]      = (_Float16)are;
    outm[row * K2_ + 64 + m] = (_Float16)aim;
}

// ---------------------------------------------------------------------------
// Phase 3: out[bho][l] = sum_k outm[bho][k] * ib[l][k]  (K = 128 -> 4 WMMAs)
// Grid: 64 row-blocks x 32 col-blocks; block 256 thr = 8 waves; wave does
// 16 rows x 128 cols (8 C tiles), reusing its A fragment across tiles.
// ib tile (128 l x 128 k, f16) async-staged in LDS, padded stride 136 halves.
// ---------------------------------------------------------------------------
#define IBS_ 136
__global__ void spectral_idft_kernel(const _Float16* __restrict__ outm,
                                     const _Float16* __restrict__ ib,
                                     float* __restrict__ out) {
    __shared__ _Float16 bt[128 * IBS_];
    const int rb   = blockIdx.x;
    const int cb   = blockIdx.y;
    const int tid  = threadIdx.x;
    const int wave = tid >> 5;
    const int lane = tid & 31;
    const int lid  = lane & 15;
    const int kb   = (lane >> 4) * 8;

    {   // cooperative 32KB async copy, 128B per thread, B128 moves
        int li = tid >> 1;
        int ks = (tid & 1) * 64;
        const _Float16* src = ib + ((size_t)(cb * 128 + li)) * K2_ + ks;
        unsigned dst = lds_off(&bt[li * IBS_ + ks]);
#pragma unroll
        for (int q = 0; q < 8; ++q)
            async_ld_b128(dst + q * 16, src + q * 8);
    }
    wait_async0();
    __syncthreads();

    const int row0 = rb * 128 + wave * 16;
    const int arow = row0 + lid;
    v8f acc[8] = {};

#pragma unroll
    for (int kk = 0; kk < 4; ++kk) {
        const _Float16* ap = outm + (size_t)arow * K2_ + kk * 32 + kb;
        v16h a = mk16(*(const v8h*)ap, *(const v8h*)(ap + 16));
#pragma unroll
        for (int t = 0; t < 8; ++t) {
            const _Float16* bp = &bt[(t * 16 + lid) * IBS_ + kk * 32 + kb];
            v16h bf = mk16(*(const v8h*)bp, *(const v8h*)(bp + 16));
            acc[t] = __builtin_amdgcn_wmma_f32_16x16x32_f16(
                false, a, false, bf, (short)0, acc[t], false, false);
        }
    }

#pragma unroll
    for (int t = 0; t < 8; ++t) {
        int col = cb * 128 + t * 16 + lid;
#pragma unroll
        for (int vg = 0; vg < 8; ++vg) {
            int row = row0 + vg + (lane >> 4) * 8;
            out[(size_t)row * L_ + col] = acc[t][vg];
        }
    }
}

// ---------------------------------------------------------------------------
extern "C" void kernel_launch(void* const* d_in, const int* in_sizes, int n_in,
                              void* d_out, int out_size, void* d_ws, size_t ws_size,
                              hipStream_t stream) {
    (void)in_sizes; (void)n_in; (void)out_size; (void)ws_size;
    const float* q      = (const float*)d_in[0];
    // d_in[1]=k, d_in[2]=v, d_in[3]=mask: unused by the reference
    const float* w_real = (const float*)d_in[4];
    const float* w_imag = (const float*)d_in[5];
    float* out = (float*)d_out;

    char* ws = (char*)d_ws;
    _Float16* fb   = (_Float16*)(ws);                        // 1 MB  [128][L]
    _Float16* ibv  = (_Float16*)(ws + (1u << 20));           // 1 MB  [L][128]
    float*    xft  = (float*)   (ws + (2u << 20));           // 4 MB  [8192][128]
    _Float16* outm = (_Float16*)(ws + (6u << 20));           // 2 MB  [8192][128]

    spectral_basis_kernel<<<2048, 256, 0, stream>>>(fb, ibv);
    spectral_dft_kernel<<<B_ * 8, 256, 0, stream>>>(q, fb, xft);
    spectral_mix_kernel<<<2048, 256, 0, stream>>>(xft, w_real, w_imag, outm);
    dim3 g3(64, 32);
    spectral_idft_kernel<<<g3, 256, 0, stream>>>(outm, ibv, out);
}
